// CausalGatedD2Attention_79224966742645
// MI455X (gfx1250) — matrix operand order repaired and checked by reading, not compiled, not run on hardware
//
#include <hip/hip_runtime.h>
#include <hip/hip_bf16.h>
#include <math.h>

#define D_MODEL 1024
#define BATCH   4
#define SEQ     4096
#define BT      (BATCH * SEQ)
#define LN_EPS  1e-5f
#define DEN_EPS 1e-6f

typedef __bf16 bf16;
typedef __attribute__((ext_vector_type(16))) __bf16 v16bf;
typedef __attribute__((ext_vector_type(8)))  __bf16 v8bf;
typedef __attribute__((ext_vector_type(8)))  float  v8f;

// ---------------------------------------------------------------------------
// WMMA fragment helpers (CDNA5 16x16x32 bf16, wave32)
// A-matrix 16x32: lane = row (M = lane&15); slots 0..7 hold K = 8*(lane>>4)..,
// slots 8..15 hold K = 16 + 8*(lane>>4).. (ISA 7.12.2).
// ---------------------------------------------------------------------------
static __device__ __forceinline__ v16bf load_a_frag(const bf16* __restrict__ mat,
                                                    size_t ld, size_t row0,
                                                    size_t k0, int lane) {
  size_t r   = row0 + (size_t)(lane & 15);
  int    grp = lane >> 4;
  const bf16* p = mat + r * ld + k0 + (size_t)(8 * grp);
  union { v16bf f; v8bf h[2]; } u;
  u.h[0] = *(const v8bf*)(p);
  u.h[1] = *(const v8bf*)(p + 16);
  return u.f;
}

// B-matrix 32x16 (KxN): lane = column (N = lane&15); slots 0..7 hold K = 8*(lane>>4)..,
// slots 8..15 hold K = 16 + 8*(lane>>4).. (same as A). Source stored [N][K].
static __device__ __forceinline__ v16bf load_b_frag(const bf16* __restrict__ mat,
                                                    size_t ld, size_t col0,
                                                    size_t k0, int lane) {
  size_t n   = col0 + (size_t)(lane & 15);
  int    grp = lane >> 4;
  const bf16* p = mat + n * ld + k0 + (size_t)(8 * grp);
  union { v16bf f; v8bf h[2]; } u;
  u.h[0] = *(const v8bf*)(p);
  u.h[1] = *(const v8bf*)(p + 16);
  return u.f;
}

static __device__ __forceinline__ v8f wmma_bf16(v16bf a, v16bf b, v8f c) {
  return __builtin_amdgcn_wmma_f32_16x16x32_bf16(false, a, false, b,
                                                 (short)0, c, false, false);
}

static __device__ __forceinline__ float elu1(float x) {
  return x > 0.f ? x + 1.f : __expf(x);
}
static __device__ __forceinline__ float sigmoidf_(float x) {
  return 1.f / (1.f + __expf(-x));
}

// ---------------------------------------------------------------------------
// LayerNorm: one block per row; emit bf16 xn.
// ---------------------------------------------------------------------------
__global__ void ln_kernel(const float* __restrict__ x,
                          const float* __restrict__ g,
                          const float* __restrict__ b,
                          bf16* __restrict__ xn) {
  int row = blockIdx.x;
  const float* xr = x + (size_t)row * D_MODEL;
  __shared__ float s1[256], s2[256];
  float lx[4];
  float sum = 0.f, sq = 0.f;
#pragma unroll
  for (int i = 0; i < 4; ++i) {
    lx[i] = xr[threadIdx.x + 256 * i];
    sum += lx[i];
    sq  += lx[i] * lx[i];
  }
  s1[threadIdx.x] = sum; s2[threadIdx.x] = sq;
  __syncthreads();
  for (int off = 128; off > 0; off >>= 1) {
    if (threadIdx.x < off) {
      s1[threadIdx.x] += s1[threadIdx.x + off];
      s2[threadIdx.x] += s2[threadIdx.x + off];
    }
    __syncthreads();
  }
  float mu  = s1[0] * (1.f / D_MODEL);
  float var = s2[0] * (1.f / D_MODEL) - mu * mu;
  float rs  = rsqrtf(var + LN_EPS);
  bf16* o = xn + (size_t)row * D_MODEL;
#pragma unroll
  for (int i = 0; i < 4; ++i) {
    int c = threadIdx.x + 256 * i;
    o[c] = (bf16)(((lx[i] - mu) * rs) * g[c] + b[c]);
  }
}

// ---------------------------------------------------------------------------
// fp32 [K,N] -> bf16 [N,K] (so WMMA B-frags load contiguous in K)
// ---------------------------------------------------------------------------
__global__ void transpose_w(const float* __restrict__ W, bf16* __restrict__ Wt,
                            int K, int N) {
  size_t total = (size_t)K * N;
  for (size_t i = (size_t)blockIdx.x * blockDim.x + threadIdx.x; i < total;
       i += (size_t)gridDim.x * blockDim.x) {
    size_t k = i / N, n = i % N;
    Wt[n * (size_t)K + k] = (bf16)W[i];
  }
}

__global__ void init_den(float* __restrict__ den) {
  int i = blockIdx.x * blockDim.x + threadIdx.x;
  if (i < BT) den[i] = DEN_EPS;
}

// ---------------------------------------------------------------------------
// q = elu(xn @ Wq + bq) + 1.  Wave: 16 rows x 64 cols; K pipelined 1-deep so
// next K-step loads overlap current WMMAs (avoids s_wait_loadcnt 0 stalls).
// ---------------------------------------------------------------------------
__global__ void gemm_q(const bf16* __restrict__ xn, const bf16* __restrict__ wt,
                       const float* __restrict__ bias, bf16* __restrict__ q) {
  int lane = threadIdx.x & 31, wave = threadIdx.x >> 5;
  int nGroup = blockIdx.x * 8 + wave;     // 16 groups of 64 cols
  size_t row0 = (size_t)blockIdx.y * 16;
  size_t n0   = (size_t)nGroup * 64;
  v8f acc[4] = {};
  v16bf aC = load_a_frag(xn, D_MODEL, row0, 0, lane);
  v16bf bC[4];
#pragma unroll
  for (int j = 0; j < 4; ++j) bC[j] = load_b_frag(wt, D_MODEL, n0 + 16 * j, 0, lane);
  for (int kk = 0; kk + 32 < D_MODEL; kk += 32) {
    v16bf aN = load_a_frag(xn, D_MODEL, row0, kk + 32, lane);
    v16bf bN[4];
#pragma unroll
    for (int j = 0; j < 4; ++j)
      bN[j] = load_b_frag(wt, D_MODEL, n0 + 16 * j, kk + 32, lane);
#pragma unroll
    for (int j = 0; j < 4; ++j) acc[j] = wmma_bf16(aC, bC[j], acc[j]);
    aC = aN;
#pragma unroll
    for (int j = 0; j < 4; ++j) bC[j] = bN[j];
  }
#pragma unroll
  for (int j = 0; j < 4; ++j) acc[j] = wmma_bf16(aC, bC[j], acc[j]);

  int col = lane & 15, grp = lane >> 4;
#pragma unroll
  for (int j = 0; j < 4; ++j) {
    size_t c = n0 + 16 * j + col;
    float bs = bias[c];
#pragma unroll
    for (int v = 0; v < 8; ++v) {
      size_t r = row0 + v + 8 * grp;
      q[r * D_MODEL + c] = (bf16)elu1(acc[j][v] + bs);
    }
  }
}

// ---------------------------------------------------------------------------
// k = elu((xn@Wk + bk) * sigmoid(xn@Wg + bg)) + 1 — dual accumulation.
// Wave: 16 rows x 32 cols (two B streams) to stay within VGPR budget.
// ---------------------------------------------------------------------------
__global__ void gemm_kg(const bf16* __restrict__ xn,
                        const bf16* __restrict__ wkT, const float* __restrict__ bk,
                        const bf16* __restrict__ wgT, const float* __restrict__ bg,
                        bf16* __restrict__ kOut) {
  int lane = threadIdx.x & 31, wave = threadIdx.x >> 5;
  int nGroup = blockIdx.x * 8 + wave;     // 32 groups of 32 cols
  size_t row0 = (size_t)blockIdx.y * 16;
  size_t n0   = (size_t)nGroup * 32;
  v8f ak[2] = {}, ag[2] = {};
  v16bf aC = load_a_frag(xn, D_MODEL, row0, 0, lane);
  v16bf kC[2], gC[2];
#pragma unroll
  for (int j = 0; j < 2; ++j) {
    kC[j] = load_b_frag(wkT, D_MODEL, n0 + 16 * j, 0, lane);
    gC[j] = load_b_frag(wgT, D_MODEL, n0 + 16 * j, 0, lane);
  }
  for (int kk = 0; kk + 32 < D_MODEL; kk += 32) {
    v16bf aN = load_a_frag(xn, D_MODEL, row0, kk + 32, lane);
    v16bf kN[2], gN[2];
#pragma unroll
    for (int j = 0; j < 2; ++j) {
      kN[j] = load_b_frag(wkT, D_MODEL, n0 + 16 * j, kk + 32, lane);
      gN[j] = load_b_frag(wgT, D_MODEL, n0 + 16 * j, kk + 32, lane);
    }
#pragma unroll
    for (int j = 0; j < 2; ++j) {
      ak[j] = wmma_bf16(aC, kC[j], ak[j]);
      ag[j] = wmma_bf16(aC, gC[j], ag[j]);
    }
    aC = aN;
#pragma unroll
    for (int j = 0; j < 2; ++j) { kC[j] = kN[j]; gC[j] = gN[j]; }
  }
#pragma unroll
  for (int j = 0; j < 2; ++j) {
    ak[j] = wmma_bf16(aC, kC[j], ak[j]);
    ag[j] = wmma_bf16(aC, gC[j], ag[j]);
  }

  int col = lane & 15, grp = lane >> 4;
#pragma unroll
  for (int j = 0; j < 2; ++j) {
    size_t c = n0 + 16 * j + col;
    float bsk = bk[c], bsg = bg[c];
#pragma unroll
    for (int v = 0; v < 8; ++v) {
      size_t r = row0 + v + 8 * grp;
      float kv = (ak[j][v] + bsk) * sigmoidf_(ag[j][v] + bsg);
      kOut[r * D_MODEL + c] = (bf16)elu1(kv);
    }
  }
}

// ---------------------------------------------------------------------------
// v = xn @ Wv + bv, stored TRANSPOSED as vT[B][D][T]
// ---------------------------------------------------------------------------
__global__ void gemm_v(const bf16* __restrict__ xn, const bf16* __restrict__ wvT,
                       const float* __restrict__ bv, bf16* __restrict__ vT) {
  int lane = threadIdx.x & 31, wave = threadIdx.x >> 5;
  int nGroup = blockIdx.x * 8 + wave;
  size_t row0 = (size_t)blockIdx.y * 16;
  size_t n0   = (size_t)nGroup * 64;
  v8f acc[4] = {};
  v16bf aC = load_a_frag(xn, D_MODEL, row0, 0, lane);
  v16bf bC[4];
#pragma unroll
  for (int j = 0; j < 4; ++j) bC[j] = load_b_frag(wvT, D_MODEL, n0 + 16 * j, 0, lane);
  for (int kk = 0; kk + 32 < D_MODEL; kk += 32) {
    v16bf aN = load_a_frag(xn, D_MODEL, row0, kk + 32, lane);
    v16bf bN[4];
#pragma unroll
    for (int j = 0; j < 4; ++j)
      bN[j] = load_b_frag(wvT, D_MODEL, n0 + 16 * j, kk + 32, lane);
#pragma unroll
    for (int j = 0; j < 4; ++j) acc[j] = wmma_bf16(aC, bC[j], acc[j]);
    aC = aN;
#pragma unroll
    for (int j = 0; j < 4; ++j) bC[j] = bN[j];
  }
#pragma unroll
  for (int j = 0; j < 4; ++j) acc[j] = wmma_bf16(aC, bC[j], acc[j]);

  int col = lane & 15, grp = lane >> 4;
#pragma unroll
  for (int j = 0; j < 4; ++j) {
    size_t c = n0 + 16 * j + col;
    float bs = bv[c];
#pragma unroll
    for (int v = 0; v < 8; ++v) {
      size_t r  = row0 + v + 8 * grp;      // global row in [0, BT)
      size_t bi = r / SEQ, tt = r % SEQ;
      vT[(bi * D_MODEL + c) * SEQ + tt] = (bf16)(acc[j][v] + bs);
    }
  }
}

// ---------------------------------------------------------------------------
// attn = tril(q @ k^T); den[t] += rowsum (atomic). Wave: 16 t-rows x 64 s-cols.
// ---------------------------------------------------------------------------
__global__ void attn_qk(const bf16* __restrict__ q, const bf16* __restrict__ k,
                        bf16* __restrict__ attn, float* __restrict__ den) {
  int lane = threadIdx.x & 31, wave = threadIdx.x >> 5;
  int sGroup = blockIdx.x * 8 + wave;     // 0..63
  int tTile  = blockIdx.y;                // 0..255
  int b      = blockIdx.z;
  int sTile0 = sGroup * 4;
  if (sTile0 > tTile) return;             // strictly above diagonal: skip
  const bf16* qb = q + (size_t)b * SEQ * D_MODEL;
  const bf16* kb = k + (size_t)b * SEQ * D_MODEL;
  bf16* ab = attn + (size_t)b * SEQ * SEQ;
  size_t row0 = (size_t)tTile * 16;
  v8f acc[4] = {};
  v16bf aC = load_a_frag(qb, D_MODEL, row0, 0, lane);
  v16bf bC[4];
#pragma unroll
  for (int j = 0; j < 4; ++j)
    bC[j] = load_b_frag(kb, D_MODEL, (size_t)(sTile0 + j) * 16, 0, lane);
  for (int kk = 0; kk + 32 < D_MODEL; kk += 32) {
    v16bf aN = load_a_frag(qb, D_MODEL, row0, kk + 32, lane);
    v16bf bN[4];
#pragma unroll
    for (int j = 0; j < 4; ++j)
      bN[j] = load_b_frag(kb, D_MODEL, (size_t)(sTile0 + j) * 16, kk + 32, lane);
#pragma unroll
    for (int j = 0; j < 4; ++j) acc[j] = wmma_bf16(aC, bC[j], acc[j]);
    aC = aN;
#pragma unroll
    for (int j = 0; j < 4; ++j) bC[j] = bN[j];
  }
#pragma unroll
  for (int j = 0; j < 4; ++j) acc[j] = wmma_bf16(aC, bC[j], acc[j]);

  int col = lane & 15, grp = lane >> 4;
  float rsum[8];
#pragma unroll
  for (int v = 0; v < 8; ++v) rsum[v] = 0.f;
#pragma unroll
  for (int j = 0; j < 4; ++j) {
    int sBase = (sTile0 + j) * 16;
#pragma unroll
    for (int v = 0; v < 8; ++v) {
      int trow = tTile * 16 + v + 8 * grp;
      int s    = sBase + col;
      float val = acc[j][v];
      if (s > trow) val = 0.f;            // causal mask
      rsum[v] += val;
      ab[(size_t)trow * SEQ + s] = (bf16)val;
    }
  }
  // butterfly reduce across the 16 lanes sharing the same rows
#pragma unroll
  for (int v = 0; v < 8; ++v) {
    rsum[v] += __shfl_xor(rsum[v], 1, 32);
    rsum[v] += __shfl_xor(rsum[v], 2, 32);
    rsum[v] += __shfl_xor(rsum[v], 4, 32);
    rsum[v] += __shfl_xor(rsum[v], 8, 32);
  }
  if ((lane & 15) == 0) {
#pragma unroll
    for (int v = 0; v < 8; ++v)
      atomicAdd(&den[(size_t)b * SEQ + tTile * 16 + v + 8 * grp], rsum[v]);
  }
}

// ---------------------------------------------------------------------------
// o = (attn @ v) / den, bf16. Wave: 16 t-rows x 64 d-cols; loops s <= t only.
// ---------------------------------------------------------------------------
__global__ void attn_av(const bf16* __restrict__ attn, const bf16* __restrict__ vT,
                        const float* __restrict__ den, bf16* __restrict__ o) {
  int lane = threadIdx.x & 31, wave = threadIdx.x >> 5;
  int dGroup = blockIdx.x * 8 + wave;     // 0..15
  int tTile  = blockIdx.y;                // 0..255
  int b      = blockIdx.z;
  const bf16* ab = attn + (size_t)b * SEQ * SEQ;
  const bf16* vb = vT + (size_t)b * D_MODEL * SEQ;
  size_t row0 = (size_t)tTile * 16;
  v8f acc[4] = {};
  int sSteps = ((tTile + 1) * 16 + 31) / 32;
  v16bf aC = load_a_frag(ab, SEQ, row0, 0, lane);
  v16bf bC[4];
#pragma unroll
  for (int j = 0; j < 4; ++j)
    bC[j] = load_b_frag(vb, SEQ, (size_t)(dGroup * 4 + j) * 16, 0, lane);
  for (int st = 0; st + 1 < sSteps; ++st) {
    int s0 = (st + 1) * 32;
    v16bf aN = load_a_frag(ab, SEQ, row0, s0, lane);
    v16bf bN[4];
#pragma unroll
    for (int j = 0; j < 4; ++j)
      bN[j] = load_b_frag(vb, SEQ, (size_t)(dGroup * 4 + j) * 16, s0, lane);
#pragma unroll
    for (int j = 0; j < 4; ++j) acc[j] = wmma_bf16(aC, bC[j], acc[j]);
    aC = aN;
#pragma unroll
    for (int j = 0; j < 4; ++j) bC[j] = bN[j];
  }
#pragma unroll
  for (int j = 0; j < 4; ++j) acc[j] = wmma_bf16(aC, bC[j], acc[j]);

  int col = lane & 15, grp = lane >> 4;
#pragma unroll
  for (int j = 0; j < 4; ++j) {
    size_t d = (size_t)dGroup * 64 + 16 * j + col;
#pragma unroll
    for (int v = 0; v < 8; ++v) {
      int trow = tTile * 16 + v + 8 * grp;
      float dn = den[(size_t)b * SEQ + trow];
      o[((size_t)b * SEQ + trow) * D_MODEL + d] = (bf16)(acc[j][v] / dn);
    }
  }
}

// ---------------------------------------------------------------------------
// out = o @ Wp + bp   (fp32 output)
// ---------------------------------------------------------------------------
__global__ void gemm_proj(const bf16* __restrict__ o, const bf16* __restrict__ wpT,
                          const float* __restrict__ bp, float* __restrict__ out) {
  int lane = threadIdx.x & 31, wave = threadIdx.x >> 5;
  int nGroup = blockIdx.x * 8 + wave;
  size_t row0 = (size_t)blockIdx.y * 16;
  size_t n0   = (size_t)nGroup * 64;
  v8f acc[4] = {};
  v16bf aC = load_a_frag(o, D_MODEL, row0, 0, lane);
  v16bf bC[4];
#pragma unroll
  for (int j = 0; j < 4; ++j) bC[j] = load_b_frag(wpT, D_MODEL, n0 + 16 * j, 0, lane);
  for (int kk = 0; kk + 32 < D_MODEL; kk += 32) {
    v16bf aN = load_a_frag(o, D_MODEL, row0, kk + 32, lane);
    v16bf bN[4];
#pragma unroll
    for (int j = 0; j < 4; ++j)
      bN[j] = load_b_frag(wpT, D_MODEL, n0 + 16 * j, kk + 32, lane);
#pragma unroll
    for (int j = 0; j < 4; ++j) acc[j] = wmma_bf16(aC, bC[j], acc[j]);
    aC = aN;
#pragma unroll
    for (int j = 0; j < 4; ++j) bC[j] = bN[j];
  }
#pragma unroll
  for (int j = 0; j < 4; ++j) acc[j] = wmma_bf16(aC, bC[j], acc[j]);

  int col = lane & 15, grp = lane >> 4;
#pragma unroll
  for (int j = 0; j < 4; ++j) {
    size_t c = n0 + 16 * j + col;
    float bs = bp[c];
#pragma unroll
    for (int v = 0; v < 8; ++v) {
      size_t r = row0 + v + 8 * grp;
      out[r * D_MODEL + c] = acc[j][v] + bs;
    }
  }
}

// ---------------------------------------------------------------------------
extern "C" void kernel_launch(void* const* d_in, const int* in_sizes, int n_in,
                              void* d_out, int out_size, void* d_ws, size_t ws_size,
                              hipStream_t stream) {
  const float* x      = (const float*)d_in[0];
  const float* w_qkv  = (const float*)d_in[1];
  const float* b_qkv  = (const float*)d_in[2];
  const float* w_gate = (const float*)d_in[3];
  const float* b_gate = (const float*)d_in[4];
  const float* w_proj = (const float*)d_in[5];
  const float* b_proj = (const float*)d_in[6];
  const float* ln_g   = (const float*)d_in[7];
  const float* ln_b   = (const float*)d_in[8];
  float* out = (float*)d_out;

  // Workspace carve-up
  char* p = (char*)d_ws;
  auto take = [&](size_t bytes) { char* q_ = p; p += (bytes + 255) & ~(size_t)255; return q_; };
  bf16*  xn     = (bf16*)take((size_t)BT * D_MODEL * 2);
  bf16*  wqkvT  = (bf16*)take((size_t)3 * D_MODEL * D_MODEL * 2);
  bf16*  wgateT = (bf16*)take((size_t)D_MODEL * D_MODEL * 2);
  bf16*  wprojT = (bf16*)take((size_t)D_MODEL * D_MODEL * 2);
  bf16*  qb     = (bf16*)take((size_t)BT * D_MODEL * 2);
  bf16*  kb     = (bf16*)take((size_t)BT * D_MODEL * 2);
  bf16*  vT     = (bf16*)take((size_t)BT * D_MODEL * 2);
  bf16*  attn   = (bf16*)take((size_t)BATCH * SEQ * SEQ * 2);
  float* den    = (float*)take((size_t)BT * 4);
  bf16*  ob     = (bf16*)take((size_t)BT * D_MODEL * 2);

  ln_kernel<<<BT, 256, 0, stream>>>(x, ln_g, ln_b, xn);
  transpose_w<<<512, 256, 0, stream>>>(w_qkv, wqkvT, D_MODEL, 3 * D_MODEL);
  transpose_w<<<256, 256, 0, stream>>>(w_gate, wgateT, D_MODEL, D_MODEL);
  transpose_w<<<256, 256, 0, stream>>>(w_proj, wprojT, D_MODEL, D_MODEL);
  init_den<<<(BT + 255) / 256, 256, 0, stream>>>(den);

  dim3 gProj(2, BT / 16);    // 16 nGroups / 8 waves-per-block, 1024 m-tiles
  gemm_q<<<gProj, 256, 0, stream>>>(xn, wqkvT, b_qkv, qb);
  dim3 gKG(4, BT / 16);      // 32 nGroups of 32 cols
  gemm_kg<<<gKG, 256, 0, stream>>>(xn, wqkvT + (size_t)D_MODEL * D_MODEL,
                                   b_qkv + D_MODEL, wgateT, b_gate, kb);
  gemm_v<<<gProj, 256, 0, stream>>>(xn, wqkvT + (size_t)2 * D_MODEL * D_MODEL,
                                    b_qkv + 2 * D_MODEL, vT);

  dim3 gQK(8, SEQ / 16, BATCH);   // 64 sGroups / 8 waves, 256 t-tiles, 4 batches
  attn_qk<<<gQK, 256, 0, stream>>>(qb, kb, attn, den);

  dim3 gAV(2, SEQ / 16, BATCH);   // 16 dGroups / 8 waves
  attn_av<<<gAV, 256, 0, stream>>>(attn, vT, den, ob);

  gemm_proj<<<gProj, 256, 0, stream>>>(ob, wprojT, b_proj, out);
}
